// UpSamplingBlock_12154757447794
// MI455X (gfx1250) — matrix operand order repaired and checked by reading, not compiled, not run on hardware
//
#include <hip/hip_runtime.h>

// Conv2d(16->64,3x3,pad1) + bias + ReLU + depth_to_space(2), fused.
// GEMM formulation: M=64 (Cout), N=256 (pixels of one row), K=144=(9 taps x 16 Cin),
// computed with V_WMMA_F32_16X16X32_F16 (f16 inputs, f32 accumulate).
// K padded to 160 with zero weights so the half-empty last K-step is harmless.

typedef __attribute__((ext_vector_type(16))) _Float16 v16h;
typedef __attribute__((ext_vector_type(8)))  _Float16 v8h;
typedef __attribute__((ext_vector_type(8)))  float    v8f;

#define XROW    258   // padded row width: col index c represents x-col (c-1), c in [0,257]
#define WSTRIDE 168   // f16 elems per Wl row (144 real + 24 zero pad); 336B stride is bank-friendly

__global__ __launch_bounds__(256)
void conv_ds2_kernel(const float* __restrict__ x,
                     const float* __restrict__ W,
                     const float* __restrict__ bias,
                     float* __restrict__ out)
{
    __shared__ __align__(32) _Float16 Xl[3 * XROW * 16];   // [r][c][ci], f16
    __shared__ __align__(32) _Float16 Wl[64 * WSTRIDE];    // [co][k], k = tap*16 + ci
    __shared__ float biasL[64];

    const int tid  = threadIdx.x;
    const int blk  = blockIdx.x;
    const int nimg = blk >> 8;     // image index (16 images)
    const int h    = blk & 255;    // conv output row

    // ---- stage input rows h-1..h+1 as f16 with ci innermost, zero-padded ----
    for (int e = tid; e < 16 * 3 * XROW; e += 256) {
        int ci  = e / (3 * XROW);
        int rem = e - ci * (3 * XROW);
        int r   = rem / XROW;
        int c   = rem - r * XROW;          // consecutive threads -> consecutive c (coalesced)
        int row = h + r - 1;
        int col = c - 1;
        float v = 0.0f;
        if ((unsigned)row < 256u && (unsigned)col < 256u)
            v = x[(((size_t)nimg * 16 + ci) * 256 + row) * 256 + col];
        Xl[(r * XROW + c) * 16 + ci] = (_Float16)v;
    }

    // ---- weights: zero the pad region k in [144,168), fill k in [0,144) ----
    for (int i = tid; i < 64 * 12; i += 256) {       // 24 f16 = 12 dwords per row
        int co  = i / 12;
        int off = i - co * 12;
        ((unsigned int*)(Wl + co * WSTRIDE + 144))[off] = 0u;
    }
    for (int i = tid; i < 64 * 144; i += 256) {
        int co  = i / 144;
        int k   = i - co * 144;
        int tap = k >> 4;                  // tap = kh*3 + kw
        int ci  = k & 15;
        Wl[co * WSTRIDE + k] = (_Float16)W[((size_t)co * 16 + ci) * 9 + tap];
    }
    if (tid < 64) biasL[tid] = bias[tid];
    __syncthreads();

    // ---- WMMA compute: each wave owns 1 M-tile x 8 N-tiles ----
    const int wid  = tid >> 5;             // wave32
    const int lane = tid & 31;
    const int lh   = lane >> 4;            // half-wave: 0 or 1
    const int lm   = lane & 15;
    const int mt   = wid & 3;              // Cout tile: channels [mt*16, mt*16+16)
    const int ntb  = (wid >> 2) * 8;       // pixel tiles [ntb, ntb+8)

    v8f acc[8] = {};

    const _Float16* wrow = &Wl[(mt * 16 + lm) * WSTRIDE];

    #pragma unroll
    for (int kk = 0; kk < 5; ++kk) {
        // A fragment (16x32 f16): per-lane K run [kb, kb+16) contiguous
        const int kb = kk * 32 + 16 * lh;
        v8h a0 = *(const v8h*)(wrow + kb);
        v8h a1 = *(const v8h*)(wrow + kb + 8);
        v16h A;
        #pragma unroll
        for (int i = 0; i < 8; ++i) { A[i] = a0[i]; A[i + 8] = a1[i]; }

        // B fragment: lane half lh covers tap t = 2*kk + lh entirely (16 contiguous ci)
        int t = 2 * kk + lh;
        if (t > 8) t = 0;                  // dummy tap: its weights are zero (k>=144)
        const int kh = t / 3;
        const int kw = t - 3 * kh;
        const _Float16* bbase = &Xl[(kh * XROW + kw + lm) * 16];

        #pragma unroll
        for (int j = 0; j < 8; ++j) {
            v16h B = *(const v16h*)(bbase + (ntb + j) * 256);  // + (nt*16 cols)*16 ci
            acc[j] = __builtin_amdgcn_wmma_f32_16x16x32_f16(
                false, A, false, B, (short)0, acc[j], false, false);
        }
    }

    // ---- epilogue: bias + ReLU + depth_to_space(2), packed float2 stores ----
    // C/D layout: VGPR r, lanes 0-15 -> M=r, lanes 16-31 -> M=r+8; N = lm.
    // ch = co*4 + dy*2 + dx; even/odd ch pair -> (dx=0,1) at same output row.
    const size_t outImg = (size_t)nimg * 16 * 512 * 512;
    #pragma unroll
    for (int j = 0; j < 8; ++j) {
        const int w2 = (((ntb + j) * 16) + lm) * 2;
        #pragma unroll
        for (int r = 0; r < 8; r += 2) {
            const int ch = mt * 16 + 8 * lh + r;   // always even
            float v0 = acc[j][r]     + biasL[ch];
            float v1 = acc[j][r + 1] + biasL[ch + 1];
            v0 = v0 > 0.0f ? v0 : 0.0f;
            v1 = v1 > 0.0f ? v1 : 0.0f;
            const int co = ch >> 2;
            const int dy = (ch >> 1) & 1;
            size_t o = outImg + ((size_t)co * 512 + (2 * h + dy)) * 512 + w2;
            float2 pv; pv.x = v0; pv.y = v1;
            *(float2*)(out + o) = pv;
        }
    }
}

extern "C" void kernel_launch(void* const* d_in, const int* in_sizes, int n_in,
                              void* d_out, int out_size, void* d_ws, size_t ws_size,
                              hipStream_t stream) {
    (void)in_sizes; (void)n_in; (void)d_ws; (void)ws_size; (void)out_size;
    const float* x  = (const float*)d_in[0];   // (16,16,256,256)
    const float* W  = (const float*)d_in[1];   // (64,16,3,3)
    const float* b  = (const float*)d_in[2];   // (64,)
    float* out = (float*)d_out;                // (16,16,512,512)
    conv_ds2_kernel<<<dim3(16 * 256), dim3(256), 0, stream>>>(x, W, b, out);
}
